// BiLinearSigmoidAttention_78417512891066
// MI455X (gfx1250) — compile-verified, matrix operand order checked by reading, not run on hardware
//
#include <hip/hip_runtime.h>
#include <hip/hip_bf16.h>
#include <math.h>

typedef __attribute__((ext_vector_type(8)))  _Float16 v8h;
typedef __attribute__((ext_vector_type(16))) _Float16 v16h;
typedef __attribute__((ext_vector_type(8)))  float    v8f;

#define S_DIM 1024
#define D_DIM 512
#define NEG_INF_F (-1e30f)

// Load 16 f16 operand values for one lane of a 16x16x32 WMMA tile.
// p points at (rowptr + ks*32 + sel*8); layout per CDNA5 ISA 7.12.2:
//   halfs [0..7] then halfs [+16 .. +23] (second K quarter for this lane group).
__device__ __forceinline__ v16h load_tile16(const _Float16* p) {
    v8h lo = *(const v8h*)(p);
    v8h hi = *(const v8h*)(p + 16);
    v16h r;
#pragma unroll
    for (int i = 0; i < 8; ++i) { r[i] = lo[i]; r[8 + i] = hi[i]; }
    return r;
}

// ---------------------------------------------------------------------------
// Kernel 1: L2-normalize rows of query and context.
//   - q-hat fp32 -> out[b,s,0:512] (first half of concat output)
//   - q-hat f16  -> ws (WMMA A operand for score GEMM)
//   - k-hat f16  -> ws (WMMA B operand for score GEMM)
// One wave32 per 512-float row (16 elems/lane).
// ---------------------------------------------------------------------------
__global__ void __launch_bounds__(256)
normalize_rows_kernel(const float* __restrict__ query,
                      const float* __restrict__ context,
                      float* __restrict__ out,       // [B,S,1024] region
                      _Float16* __restrict__ qh,     // [B,S,512] f16
                      _Float16* __restrict__ kh,     // [B,S,512] f16
                      int B)
{
    const int gwave = (int)((blockIdx.x * blockDim.x + threadIdx.x) >> 5);
    const int lane  = (int)(threadIdx.x & 31);
    const int rows  = B * S_DIM;
    if (gwave >= 2 * rows) return;

    const bool is_q = gwave < rows;
    const int  row  = is_q ? gwave : gwave - rows;
    const float* src = (is_q ? query : context) + (size_t)row * D_DIM;

    float v[16];
    float ss = 0.f;
#pragma unroll
    for (int i = 0; i < 16; ++i) {
        v[i] = src[lane + i * 32];
        ss += v[i] * v[i];
    }
#pragma unroll
    for (int off = 16; off >= 1; off >>= 1)
        ss += __shfl_xor(ss, off, 32);

    const float nrm = sqrtf(ss);
    const float inv = (nrm == 0.f) ? 1.f : 1.f / nrm;   // norm==0 -> 1 (ref semantics)

    _Float16* hdst = (is_q ? qh : kh) + (size_t)row * D_DIM;
    float*    fdst = out + (size_t)row * 1024;          // only used when is_q
#pragma unroll
    for (int i = 0; i < 16; ++i) {
        const float nv = v[i] * inv;
        hdst[lane + i * 32] = (_Float16)nv;
        if (is_q) fdst[lane + i * 32] = nv;
    }
}

// ---------------------------------------------------------------------------
// Kernel 1b: transpose context [B,S,D] fp32 -> ctxT [B,D,S] f16 so the
// attended GEMM's B operand is contiguous along the contraction dim.
// Classic 32x32 tile through padded LDS; both global accesses coalesced.
// ---------------------------------------------------------------------------
__global__ void __launch_bounds__(256)
transpose_ctx_kernel(const float* __restrict__ context,
                     _Float16* __restrict__ ctxT,
                     int B)
{
    __shared__ float tile[32][33];                      // +1 pad: bank-conflict free
    const int blocksPerB = (S_DIM / 32) * (D_DIM / 32); // 32 * 16 = 512
    const int b  = (int)blockIdx.x / blocksPerB;
    const int r  = (int)blockIdx.x % blocksPerB;
    const int st = r / (D_DIM / 32);
    const int dt = r % (D_DIM / 32);
    const int sbase = st * 32, dbase = dt * 32;

    const float* src = context + (size_t)b * S_DIM * D_DIM;
    _Float16*    dst = ctxT    + (size_t)b * D_DIM * S_DIM;

#pragma unroll
    for (int i = 0; i < 4; ++i) {
        const int e  = (int)threadIdx.x + i * 256;
        const int rr = e >> 5, cc = e & 31;             // rr = s offset, cc = d offset
        tile[rr][cc] = src[(size_t)(sbase + rr) * D_DIM + dbase + cc];
    }
    __syncthreads();
#pragma unroll
    for (int i = 0; i < 4; ++i) {
        const int e  = (int)threadIdx.x + i * 256;
        const int rr = e >> 5, cc = e & 31;             // rr = d offset, cc = s offset
        dst[(size_t)(dbase + rr) * S_DIM + sbase + cc] = (_Float16)tile[cc][rr];
    }
}

// ---------------------------------------------------------------------------
// Kernel 2: fused scores + L1 norm + attended-context.
// One block per (batch, 16-row query tile); 8 waves (256 threads).
// ---------------------------------------------------------------------------
__global__ void __launch_bounds__(256)
bilinear_sigmoid_attn_kernel(const _Float16* __restrict__ qh,     // [B,S,512] f16 (normalized)
                             const _Float16* __restrict__ kh,     // [B,S,512] f16 (normalized)
                             const _Float16* __restrict__ ctxT,   // [B,512,S] f16 (raw context^T)
                             const int* __restrict__ length,      // [B]
                             float* __restrict__ out,             // [B,S,1024]
                             float* __restrict__ scores_out,      // [B,S,1024]
                             int B)
{
    __shared__ __align__(16) _Float16 s_scores[16 * S_DIM];   // 32 KB, scores tile (f16)
    __shared__ float s_rs[8][16];                             // per-wave row sums
    __shared__ float s_scale[16];                             // final per-row scale

    const int b     = (int)(blockIdx.x) / (S_DIM / 16);
    const int qt    = (int)(blockIdx.x) % (S_DIM / 16);
    const int qbase = qt * 16;
    const int wv    = (int)(threadIdx.x >> 5);
    const int lane  = (int)(threadIdx.x & 31);
    const int hrow  = lane & 15;   // M row (A/C) or N column (B/C) within tile
    const int sel   = lane >> 4;   // which K-half this lane group holds
    const int len   = length[b];

    // ---------------- Phase A: raw scores -> sigmoid -> LDS (pre-norm) ------
    const _Float16* arow = qh + ((size_t)b * S_DIM + qbase + hrow) * D_DIM;
    float rs[8];
#pragma unroll
    for (int i = 0; i < 8; ++i) rs[i] = 0.f;

    for (int nt = wv * 8; nt < wv * 8 + 8; ++nt) {      // 8 n-tiles per wave
        const int ncol0 = nt * 16;
        const _Float16* brow = kh + ((size_t)b * S_DIM + ncol0 + hrow) * D_DIM;
        v8f c = {};
        for (int ks = 0; ks < 16; ++ks) {               // K = 512 = 16 x 32
            v16h a  = load_tile16(arow + ks * 32 + sel * 8);
            v16h bm = load_tile16(brow + ks * 32 + sel * 8);
            c = __builtin_amdgcn_wmma_f32_16x16x32_f16(
                    false, a, false, bm, (short)0, c, false, false);
        }
        const int  ncol  = ncol0 + hrow;
        const bool valid = ncol < len;
#pragma unroll
        for (int i = 0; i < 8; ++i) {
            const float raw = valid ? c[i] : NEG_INF_F;
            const float sg  = 1.f / (1.f + __expf(-raw));   // sigmoid; ==0 when masked
            rs[i] += sg;
            const int m = i + sel * 8;                      // C layout: VGPR i -> M=i (+8 hi lanes)
            s_scores[m * S_DIM + ncol] = (_Float16)sg;
        }
    }

    // Half-wave reduce keeps M=0..7 (lanes 0-15) and M=8..15 (lanes 16-31) separate.
#pragma unroll
    for (int i = 0; i < 8; ++i) {
#pragma unroll
        for (int off = 1; off < 16; off <<= 1)
            rs[i] += __shfl_xor(rs[i], off, 32);
    }
    if (hrow == 0) {
#pragma unroll
        for (int i = 0; i < 8; ++i) s_rs[wv][i + sel * 8] = rs[i];
    }
    __syncthreads();

    // ---------------- Phase B: per-row L1 scale + query-row mask ------------
    if (threadIdx.x < 16) {
        float t = 0.f;
#pragma unroll
        for (int w = 0; w < 8; ++w) t += s_rs[w][threadIdx.x];
        float sc = (t < 1.f) ? 1.f : 1.f / t;               // divide only if norm >= 1
        if (qbase + (int)threadIdx.x >= len) sc = 0.f;      // zero masked query rows
        s_scale[threadIdx.x] = sc;
    }
    __syncthreads();

    // ---------------- Phase C: write normalized scores, rescale LDS in place
    {
        float* srow = scores_out + ((size_t)b * S_DIM + qbase) * S_DIM;
        for (int idx = (int)threadIdx.x; idx < 16 * S_DIM; idx += 256) {
            const int m   = idx >> 10;
            const int col = idx & (S_DIM - 1);
            const float sval = (float)s_scores[idx] * s_scale[m];
            srow[(size_t)m * S_DIM + col] = sval;
            s_scores[idx] = (_Float16)sval;                 // A operand for phase D
        }
    }
    __syncthreads();

    // ---------------- Phase D: attended = scores (16x1024) x context (1024x512)
    // B operand now contiguous thanks to ctxT: same load pattern as phase A.
    const _Float16* ctxT_b = ctxT + (size_t)b * D_DIM * S_DIM;
    const int nb = wv * 64;                                 // this wave's D-column slice
    for (int nt = 0; nt < 4; ++nt) {
        const int n = nb + nt * 16 + hrow;                  // B/C column for this lane
        const _Float16* brow = ctxT_b + (size_t)n * S_DIM;  // ctxT row n: k-contiguous
        v8f acc = {};
        for (int ks = 0; ks < 32; ++ks) {                   // K = Sc = 1024 = 32 x 32
            v16h a  = load_tile16(&s_scores[hrow * S_DIM + ks * 32 + sel * 8]);
            v16h bm = load_tile16(brow + ks * 32 + sel * 8);
            acc = __builtin_amdgcn_wmma_f32_16x16x32_f16(
                      false, a, false, bm, (short)0, acc, false, false);
        }
#pragma unroll
        for (int i = 0; i < 8; ++i) {
            const int m = i + sel * 8;
            out[((size_t)b * S_DIM + qbase + m) * 1024 + D_DIM + n] = acc[i];
        }
    }
}

// ---------------------------------------------------------------------------
extern "C" void kernel_launch(void* const* d_in, const int* in_sizes, int n_in,
                              void* d_out, int out_size, void* d_ws, size_t ws_size,
                              hipStream_t stream) {
    const float* context = (const float*)d_in[0];
    const float* query   = (const float*)d_in[1];
    const int*   length  = (const int*)d_in[2];
    const int B = in_sizes[2];                      // 32 in the reference setup

    float* out_main   = (float*)d_out;                                  // [B,S,1024]
    float* out_scores = (float*)d_out + (size_t)B * S_DIM * 1024;       // [B,S,1024]

    _Float16* qh   = (_Float16*)d_ws;                                   // B*S*512 f16
    _Float16* kh   = qh + (size_t)B * S_DIM * D_DIM;                    // B*S*512 f16
    _Float16* ctxT = kh + (size_t)B * S_DIM * D_DIM;                    // B*512*S f16

    // Kernel 1: one wave per row, 8 waves per block.
    {
        const int rows   = 2 * B * S_DIM;
        const int blocks = (rows + 7) / 8;
        normalize_rows_kernel<<<blocks, 256, 0, stream>>>(
            query, context, out_main, qh, kh, B);
    }
    // Kernel 1b: context -> f16 transpose for the attended GEMM.
    {
        const int blocks = B * (S_DIM / 32) * (D_DIM / 32);
        transpose_ctx_kernel<<<blocks, 256, 0, stream>>>(context, ctxT, B);
    }
    // Kernel 2: one block per (b, 16-row q tile).
    {
        const int blocks = B * (S_DIM / 16);
        bilinear_sigmoid_attn_kernel<<<blocks, 256, 0, stream>>>(
            qh, kh, ctxT, length, out_main, out_scores, B);
    }
}
